// GalerkinAttention_44384192037438
// MI455X (gfx1250) — matrix operand order repaired and checked by reading, not compiled, not run on hardware
//
#include <hip/hip_runtime.h>
#include <hip/hip_bf16.h>

// ---------------- CDNA5 WMMA plumbing ----------------
typedef __attribute__((ext_vector_type(16))) __bf16 v16bf;
typedef __attribute__((ext_vector_type(8)))  float  v8f;

union FragU { unsigned int u[8]; v16bf bf; };

// f32 -> packed bf16x2 with round-to-nearest-even via the hardware converter.
// D[15:0] = bf16(a), D[31:16] = bf16(b)
__device__ __forceinline__ unsigned int pack2(float a, float b) {
  unsigned int r;
  asm("v_cvt_pk_bf16_f32 %0, %1, %2" : "=v"(r) : "v"(a), "v"(b));
  return r;
}
__device__ __forceinline__ unsigned short f2bf(float x) {
  return (unsigned short)pack2(x, x);
}

// A fragment (16x32 bf16), ISA 7.12.2: lane holds row M=lane%16;
// half=lane/16: Vj -> K = half*16 + 2j,2j+1 (code's contiguous K order, not the ISA table's; same as B)
__device__ __forceinline__ void fragA_lds(const unsigned short* p, int row, int ldk, FragU& f) {
  const int hf = (threadIdx.x >> 4) & 1;
  const unsigned short* r = p + row * ldk;
#pragma unroll
  for (int j = 0; j < 4; ++j) {
    const int k0 = hf * 16 + 2 * j;
    f.u[j]     = *(const unsigned int*)(r + k0);
    f.u[j + 4] = *(const unsigned int*)(r + k0 + 8);
  }
}
// B fragment (32x16 bf16): lane holds col N=lane%16; Vj -> K = (lane/16)*16 + 2j,2j+1
__device__ __forceinline__ void fragB_lds(const unsigned short* p, int row, int ldk, FragU& f) {
  const int hf = (threadIdx.x >> 4) & 1;
  const unsigned short* r = p + row * ldk + hf * 16;
#pragma unroll
  for (int j = 0; j < 8; ++j) f.u[j] = *(const unsigned int*)(r + 2 * j);
}
__device__ __forceinline__ v8f wmma_bf(const FragU& a, const FragU& b, v8f c) {
  return __builtin_amdgcn_wmma_f32_16x16x32_bf16(false, a.bf, false, b.bf, (short)0, c,
                                                 false, false);
}

// problem constants (fixed by setup_inputs)
#define NB    2
#define NN    32768
#define DIMI  512
#define NH    8
#define DH    64
#define NG    32
#define BN    65536   /* NB*NN */
#define TRI   1536    /* 3*DIMI */

// ---------------- K1: group starts + inverse sizes ----------------
__global__ void ga_groups(const int* __restrict__ batch, int* __restrict__ starts,
                          float* __restrict__ inv) {
  __shared__ int cnt[NG];
  const int t = threadIdx.x;
  if (t < NG) cnt[t] = 0;
  __syncthreads();
  for (int i = t; i < NN; i += 256) atomicAdd(&cnt[batch[i]], 1);
  __syncthreads();
  if (t == 0) {
    int acc = 0;
    for (int g = 0; g < NG; ++g) { starts[g] = acc; acc += cnt[g]; }
    starts[NG] = acc;
  }
  if (t < NG) inv[t] = cnt[t] ? 1.0f / (float)cnt[t] : 0.0f;
}

// ---------------- K2: qkv GEMM + fused LN / q-scale, bf16 repack ----------------
// block tile: 128 rows x 64 cols (one head slice), 8 waves x (16x64)
__global__ __launch_bounds__(256) void ga_qkv(
    const float* __restrict__ x, const float* __restrict__ w_qkv,
    const float* __restrict__ ln1w, const float* __restrict__ ln1b,
    const float* __restrict__ ln2w, const float* __restrict__ ln2b,
    const int* __restrict__ batch, const float* __restrict__ inv,
    unsigned short* __restrict__ qbf, unsigned short* __restrict__ kT,
    unsigned short* __restrict__ vT) {
  __shared__ alignas(16) unsigned short la[128 * 32];
  __shared__ alignas(16) unsigned short lb[64 * 32];
  const int t = threadIdx.x;
  const int wave = t >> 5, hf = (t >> 4) & 1, l16 = t & 15;
  const int c0 = blockIdx.x * 64;    // output col base in [0,1536)
  const int m0 = blockIdx.y * 128;   // flattened row base in [0,65536)

  const int arow = t >> 1, acol = (t & 1) * 16;   // A staging: 16 f32 each
  const int brow = t >> 2, bcol = (t & 3) * 8;    // B staging: 8 f32 each

  v8f z = {};
  v8f acc[4] = {z, z, z, z};

  for (int k0 = 0; k0 < DIMI; k0 += 32) {
    {  // stage X tile (f32 -> bf16 via v_cvt_pk_bf16_f32)
      const float* s = x + (size_t)(m0 + arow) * DIMI + k0 + acol;
      float4 f0 = *(const float4*)(s + 0), f1 = *(const float4*)(s + 4);
      float4 f2 = *(const float4*)(s + 8), f3 = *(const float4*)(s + 12);
      unsigned int* d = (unsigned int*)(la + arow * 32 + acol);
      d[0] = pack2(f0.x, f0.y); d[1] = pack2(f0.z, f0.w);
      d[2] = pack2(f1.x, f1.y); d[3] = pack2(f1.z, f1.w);
      d[4] = pack2(f2.x, f2.y); d[5] = pack2(f2.z, f2.w);
      d[6] = pack2(f3.x, f3.y); d[7] = pack2(f3.z, f3.w);
      if (k0 + 32 < DIMI) __builtin_prefetch(s + 32, 0, 0);
    }
    {  // stage W tile (rows of w_qkv are output cols)
      const float* s = w_qkv + (size_t)(c0 + brow) * DIMI + k0 + bcol;
      float4 f0 = *(const float4*)(s + 0), f1 = *(const float4*)(s + 4);
      unsigned int* d = (unsigned int*)(lb + brow * 32 + bcol);
      d[0] = pack2(f0.x, f0.y); d[1] = pack2(f0.z, f0.w);
      d[2] = pack2(f1.x, f1.y); d[3] = pack2(f1.z, f1.w);
    }
    __syncthreads();
    FragU a, b;
    fragA_lds(la, wave * 16 + l16, 32, a);
#pragma unroll
    for (int tn = 0; tn < 4; ++tn) {
      fragB_lds(lb, tn * 16 + l16, 32, b);
      acc[tn] = wmma_bf(a, b, acc[tn]);
    }
    __syncthreads();
  }

  const int headg = c0 >> 6;       // 0..23
  const int which = headg >> 3;    // 0=q 1=k 2=v
  const int h = headg & 7;

  if (which == 0) {  // q: scale by 1/size(group)
#pragma unroll
    for (int v = 0; v < 8; ++v) {
      const int m = m0 + wave * 16 + v + hf * 8;
      const int b_ = m >> 15, n = m & (NN - 1);
      const float iv = inv[batch[n]];
      unsigned short* d = qbf + ((size_t)(b_ * NH + h) * NN + n) * DH;
#pragma unroll
      for (int tn = 0; tn < 4; ++tn) d[tn * 16 + l16] = f2bf(acc[tn][v] * iv);
    }
  } else {           // k/v: layernorm over the 64 cols of this head
    const float* lw = (which == 1) ? ln1w : ln2w;
    const float* lbp = (which == 1) ? ln1b : ln2b;
    unsigned short* outT = (which == 1) ? kT : vT;
    float w4[4], b4[4];
#pragma unroll
    for (int tn = 0; tn < 4; ++tn) { w4[tn] = lw[tn * 16 + l16]; b4[tn] = lbp[tn * 16 + l16]; }
#pragma unroll
    for (int v = 0; v < 8; ++v) {
      float s = 0.f, s2 = 0.f;
#pragma unroll
      for (int tn = 0; tn < 4; ++tn) { const float xv = acc[tn][v]; s += xv; s2 += xv * xv; }
#pragma unroll
      for (int off = 1; off < 16; off <<= 1) {  // 16-lane group reduce (wave32)
        s += __shfl_xor(s, off, 32);
        s2 += __shfl_xor(s2, off, 32);
      }
      const float mu = s * (1.0f / 64.0f);
      const float var = s2 * (1.0f / 64.0f) - mu * mu;
      const float rs = rsqrtf(var + 1e-6f);
      const int m = m0 + wave * 16 + v + hf * 8;
      const int b_ = m >> 15, n = m & (NN - 1);
      unsigned short* d = outT + (size_t)((b_ * NH + h) * DH) * NN;
#pragma unroll
      for (int tn = 0; tn < 4; ++tn) {
        const int col = tn * 16 + l16;
        d[(size_t)col * NN + n] = f2bf((acc[tn][v] - mu) * rs * w4[tn] + b4[tn]);
      }
    }
  }
}

// ---------------- K3: ktv[g] = K[s:e]^T V[s:e]  (64x64 out, contiguous groups) ---
__global__ __launch_bounds__(256) void ga_ktv(const unsigned short* __restrict__ kT,
                                              const unsigned short* __restrict__ vT,
                                              const int* __restrict__ starts,
                                              unsigned short* __restrict__ ktvT) {
  __shared__ alignas(16) unsigned short lk[64 * 32];
  __shared__ alignas(16) unsigned short lv[64 * 32];
  const int g = blockIdx.x, bh = blockIdx.y;
  const int t = threadIdx.x, wave = t >> 5, hf = (t >> 4) & 1, l16 = t & 15;
  const int s = starts[g], e = starts[g + 1];
  const unsigned short* kb = kT + (size_t)bh * DH * NN;
  const unsigned short* vb = vT + (size_t)bh * DH * NN;
  v8f z = {};
  v8f a0 = z, a1 = z;
  const int mt = wave & 3, nt = (wave >> 2) * 2;   // 4x2 wave grid, 16x32 each
  const int srow = t >> 2, scol = (t & 3) * 8;
  for (int n0 = s & ~31; n0 < e; n0 += 32) {
#pragma unroll
    for (int j = 0; j < 8; ++j) {
      const int n = n0 + scol + j;
      const bool ok = (n >= s) && (n < e);
      lk[srow * 32 + scol + j] = ok ? kb[(size_t)srow * NN + n] : (unsigned short)0;
      lv[srow * 32 + scol + j] = ok ? vb[(size_t)srow * NN + n] : (unsigned short)0;
    }
    __syncthreads();
    FragU a, b;
    fragA_lds(lk, mt * 16 + l16, 32, a);
    fragB_lds(lv, nt * 16 + l16, 32, b);       a0 = wmma_bf(a, b, a0);
    fragB_lds(lv, (nt + 1) * 16 + l16, 32, b); a1 = wmma_bf(a, b, a1);
    __syncthreads();
  }
  // store transposed: ktvT[g][bh][dv][dk], bf16
  unsigned short* d = ktvT + (size_t)(g * (NB * NH) + bh) * (DH * DH);
#pragma unroll
  for (int v = 0; v < 8; ++v) {
    const int m = mt * 16 + v + hf * 8;      // dk
    const int c = nt * 16 + l16;             // dv
    d[(size_t)c * DH + m] = f2bf(a0[v]);
    d[(size_t)(c + 16) * DH + m] = f2bf(a1[v]);
  }
}

// ---------------- K4: og[n] = qn[n] * ktv[batch[n]], write attn bf16 ------------
__global__ __launch_bounds__(256) void ga_apply(const unsigned short* __restrict__ qbf,
                                                const unsigned short* __restrict__ ktvT,
                                                const int* __restrict__ batch,
                                                unsigned short* __restrict__ attn) {
  const int t = threadIdx.x, wave = t >> 5, hf = (t >> 4) & 1, l16 = t & 15;
  const int nt128 = blockIdx.x & 255;
  const int bh = blockIdx.x >> 8;
  const int r0 = nt128 * 128 + wave * 16;
  const int nrow = r0 + l16;
  const int grow = batch[nrow];
  const int gmin = __shfl(grow, 0, 32);
  const int gmax = __shfl(grow, 15, 32);   // batch is sorted
  v8f z = {};
  v8f acc[4] = {z, z, z, z};
  const unsigned short* qrow = qbf + ((size_t)bh * NN + nrow) * DH;
  for (int g = gmin; g <= gmax; ++g) {     // usually a single iteration
    const bool msk = (grow == g);
    const unsigned short* kb = ktvT + (size_t)(g * (NB * NH) + bh) * (DH * DH);
#pragma unroll
    for (int kc = 0; kc < 2; ++kc) {
      FragU a;
#pragma unroll
      for (int j = 0; j < 4; ++j) {
        const int k0 = kc * 32 + hf * 16 + 2 * j;
        a.u[j]     = msk ? *(const unsigned int*)(qrow + k0) : 0u;
        a.u[j + 4] = msk ? *(const unsigned int*)(qrow + k0 + 8) : 0u;
      }
#pragma unroll
      for (int tn = 0; tn < 4; ++tn) {
        FragU b;
        const unsigned short* br = kb + (size_t)(tn * 16 + l16) * DH + kc * 32 + hf * 16;
#pragma unroll
        for (int j = 0; j < 8; ++j) b.u[j] = *(const unsigned int*)(br + 2 * j);
        acc[tn] = wmma_bf(a, b, acc[tn]);
      }
    }
  }
  const int b_ = bh >> 3, h = bh & 7;
#pragma unroll
  for (int v = 0; v < 8; ++v) {
    const int n = r0 + v + hf * 8;
    unsigned short* d = attn + ((size_t)b_ * NN + n) * DIMI + h * DH;
#pragma unroll
    for (int tn = 0; tn < 4; ++tn) d[tn * 16 + l16] = f2bf(acc[tn][v]);
  }
}

// ---------------- K5: out = attn @ w_out^T + b_out (f32 out) --------------------
__global__ __launch_bounds__(256) void ga_proj(const unsigned short* __restrict__ attn,
                                               const float* __restrict__ w_out,
                                               const float* __restrict__ b_out,
                                               float* __restrict__ out) {
  __shared__ alignas(16) unsigned short la[128 * 32];
  __shared__ alignas(16) unsigned short lb[64 * 32];
  const int t = threadIdx.x, wave = t >> 5, hf = (t >> 4) & 1, l16 = t & 15;
  const int c0 = blockIdx.x * 64, m0 = blockIdx.y * 128;
  v8f z = {};
  v8f acc[4] = {z, z, z, z};
  const int arow = t >> 1, acol = (t & 1) * 16;
  const int brow = t >> 2, bcol = (t & 3) * 8;
  for (int k0 = 0; k0 < DIMI; k0 += 32) {
    {  // A already bf16: straight 2x b128 copies
      const uint4* s = (const uint4*)(attn + (size_t)(m0 + arow) * DIMI + k0 + acol);
      uint4* d = (uint4*)(la + arow * 32 + acol);
      d[0] = s[0]; d[1] = s[1];
      if (k0 + 32 < DIMI)
        __builtin_prefetch(attn + (size_t)(m0 + arow) * DIMI + k0 + 32 + acol, 0, 0);
    }
    {  // B: w_out rows = output cols, f32 -> bf16
      const float* s = w_out + (size_t)(c0 + brow) * DIMI + k0 + bcol;
      float4 f0 = *(const float4*)(s + 0), f1 = *(const float4*)(s + 4);
      unsigned int* d = (unsigned int*)(lb + brow * 32 + bcol);
      d[0] = pack2(f0.x, f0.y); d[1] = pack2(f0.z, f0.w);
      d[2] = pack2(f1.x, f1.y); d[3] = pack2(f1.z, f1.w);
    }
    __syncthreads();
    FragU a, b;
    fragA_lds(la, wave * 16 + l16, 32, a);
#pragma unroll
    for (int tn = 0; tn < 4; ++tn) {
      fragB_lds(lb, tn * 16 + l16, 32, b);
      acc[tn] = wmma_bf(a, b, acc[tn]);
    }
    __syncthreads();
  }
  float bias[4];
#pragma unroll
  for (int tn = 0; tn < 4; ++tn) bias[tn] = b_out[c0 + tn * 16 + l16];
#pragma unroll
  for (int v = 0; v < 8; ++v) {
    const int m = m0 + wave * 16 + v + hf * 8;
    float* d = out + (size_t)m * DIMI + c0;
#pragma unroll
    for (int tn = 0; tn < 4; ++tn) d[tn * 16 + l16] = acc[tn][v] + bias[tn];
  }
}

// ---------------- launcher ----------------
extern "C" void kernel_launch(void* const* d_in, const int* in_sizes, int n_in,
                              void* d_out, int out_size, void* d_ws, size_t ws_size,
                              hipStream_t stream) {
  (void)in_sizes; (void)n_in; (void)out_size; (void)ws_size;
  const float* x     = (const float*)d_in[0];
  const float* w_qkv = (const float*)d_in[1];
  const float* ln1w  = (const float*)d_in[2];
  const float* ln1b  = (const float*)d_in[3];
  const float* ln2w  = (const float*)d_in[4];
  const float* ln2b  = (const float*)d_in[5];
  const float* w_out = (const float*)d_in[6];
  const float* b_out = (const float*)d_in[7];
  const int*   batch = (const int*)d_in[8];
  float* out = (float*)d_out;

  char* ws = (char*)d_ws;
  int*   starts = (int*)ws;              // 33 ints
  float* inv    = (float*)(ws + 256);    // 32 floats
  unsigned short* qbf  = (unsigned short*)(ws + 1024);            // [B][H][N][D] bf16
  unsigned short* kT   = qbf + (size_t)BN * DIMI;                 // [B][H][D][N] bf16
  unsigned short* vT   = kT + (size_t)BN * DIMI;                  // [B][H][D][N] bf16
  unsigned short* ktvT = vT + (size_t)BN * DIMI;                  // [G][BH][D][D] bf16
  unsigned short* attn = kT;  // alias: kT is dead after ga_ktv

  ga_groups<<<1, 256, 0, stream>>>(batch, starts, inv);
  ga_qkv<<<dim3(TRI / 64, BN / 128), 256, 0, stream>>>(x, w_qkv, ln1w, ln1b, ln2w, ln2b,
                                                       batch, inv, qbf, kT, vT);
  ga_ktv<<<dim3(NG, NB * NH), 256, 0, stream>>>(kT, vT, starts, ktvT);
  ga_apply<<<NB * NH * (NN / 128), 256, 0, stream>>>(qbf, ktvT, batch, attn);
  ga_proj<<<dim3(DIMI / 64, BN / 128), 256, 0, stream>>>(attn, w_out, b_out, out);
}
